// FocusedLSTM_28020366639386
// MI455X (gfx1250) — compile-verified, matrix operand order checked by reading, not run on hardware
//
#include <hip/hip_runtime.h>
#include <hip/hip_bf16.h>

typedef __attribute__((ext_vector_type(16))) __bf16 v16bf;
typedef __attribute__((ext_vector_type(8)))  __bf16 v8bf;
typedef __attribute__((ext_vector_type(8)))  float  v8f;

__device__ __forceinline__ unsigned short f2bf_rne(float f) {
  unsigned u = __float_as_uint(f);
  u += 0x7FFFu + ((u >> 16) & 1u);   // round-to-nearest-even to bf16
  return (unsigned short)(u >> 16);
}
__device__ __forceinline__ __bf16 us2bf(unsigned short h) {
  return __builtin_bit_cast(__bf16, h);
}

// ---------------------------------------------------------------------------
// Kernel 1: pack R (f32 [512][1024]) -> bf16 in WMMA B-fragment order.
// Rpk[((kt*64 + nt)*32 + lane)*16 + 2v + {0,1}] =
//   bf16(R[kt*32 + kb (+1)][nt*16 + (lane&15)]),  kb=(v>>2)*16+(lane>>4)*8+(v&3)*2
// ---------------------------------------------------------------------------
__global__ __launch_bounds__(256) void pack_R_kernel(
    const float* __restrict__ R, unsigned short* __restrict__ Rpk) {
  const int gid  = blockIdx.x * 256 + threadIdx.x;   // == (kt*64+nt)*32+lane
  const int kt   = gid >> 11;
  const int rem  = gid & 2047;
  const int nt   = rem >> 5;
  const int lane = rem & 31;
  const int n    = nt * 16 + (lane & 15);
  const int kh   = (lane >> 4) * 8;
  unsigned short* out = Rpk + (size_t)gid * 16;
#pragma unroll
  for (int v = 0; v < 8; ++v) {
    const int kb = (v >> 2) * 16 + kh + (v & 3) * 2;
    const int k  = kt * 32 + kb;
    out[2 * v]     = f2bf_rne(R[(size_t)k * 1024 + n]);
    out[2 * v + 1] = f2bf_rne(R[(size_t)(k + 1) * 1024 + n]);
  }
}

// ---------------------------------------------------------------------------
// Kernel 2: z[t][b][h] = sigmoid(x@W + b_w).  Block = 256 thr (8 waves),
// 64x64 output tile, K staged 32-wide through LDS in bf16, WMMA bf16->f32.
// ---------------------------------------------------------------------------
__global__ __launch_bounds__(256) void z_gemm_kernel(
    const float* __restrict__ x, const float* __restrict__ W,
    const float* __restrict__ b_w, float* __restrict__ z) {
  __shared__ unsigned short As[64][32];    // [m][k]  bf16
  __shared__ unsigned short BsT[64][32];   // [n][k]  bf16 (W transposed)
  const int tid  = threadIdx.x;
  const int lane = tid & 31;
  const int wv   = tid >> 5;               // 0..7
  const int m0   = blockIdx.y * 64;
  const int n0   = blockIdx.x * 64;
  const int tm   = wv >> 1;                // 0..3  (row tile)
  const int tn0  = (wv & 1) * 2;           // {0,2} (2 col tiles per wave)

  v8f zero = {};
  v8f acc0 = zero, acc1 = zero;

  const int arow = tid >> 2, achunk = (tid & 3) * 8;  // A staging: 8 f32/thread
  const int bk   = tid >> 3, bn     = (tid & 7) * 8;  // B staging: 8 f32/thread

  for (int kt = 0; kt < 16; ++kt) {
    const int k0 = kt * 32;
    {
      const float* p = x + (size_t)(m0 + arow) * 512 + k0 + achunk;
      float4 v0 = ((const float4*)p)[0];
      float4 v1 = ((const float4*)p)[1];
      As[arow][achunk + 0] = f2bf_rne(v0.x); As[arow][achunk + 1] = f2bf_rne(v0.y);
      As[arow][achunk + 2] = f2bf_rne(v0.z); As[arow][achunk + 3] = f2bf_rne(v0.w);
      As[arow][achunk + 4] = f2bf_rne(v1.x); As[arow][achunk + 5] = f2bf_rne(v1.y);
      As[arow][achunk + 6] = f2bf_rne(v1.z); As[arow][achunk + 7] = f2bf_rne(v1.w);
    }
    {
      const float* p = W + (size_t)(k0 + bk) * 512 + n0 + bn;
      float4 v0 = ((const float4*)p)[0];
      float4 v1 = ((const float4*)p)[1];
      BsT[bn + 0][bk] = f2bf_rne(v0.x); BsT[bn + 1][bk] = f2bf_rne(v0.y);
      BsT[bn + 2][bk] = f2bf_rne(v0.z); BsT[bn + 3][bk] = f2bf_rne(v0.w);
      BsT[bn + 4][bk] = f2bf_rne(v1.x); BsT[bn + 5][bk] = f2bf_rne(v1.y);
      BsT[bn + 6][bk] = f2bf_rne(v1.z); BsT[bn + 7][bk] = f2bf_rne(v1.w);
    }
    __syncthreads();

    v16bf a, b0, b1;
    const int mrow = tm * 16 + (lane & 15);
    const int nl0  = tn0 * 16 + (lane & 15);
    const int nl1  = (tn0 + 1) * 16 + (lane & 15);
    const int kh   = (lane >> 4) * 8;
#pragma unroll
    for (int v = 0; v < 8; ++v) {
      const int kb = (v >> 2) * 16 + kh + (v & 3) * 2;
      unsigned pa = *(const unsigned*)&As[mrow][kb];
      a[2 * v]     = us2bf((unsigned short)(pa & 0xFFFFu));
      a[2 * v + 1] = us2bf((unsigned short)(pa >> 16));
      unsigned p0 = *(const unsigned*)&BsT[nl0][kb];
      b0[2 * v]     = us2bf((unsigned short)(p0 & 0xFFFFu));
      b0[2 * v + 1] = us2bf((unsigned short)(p0 >> 16));
      unsigned p1 = *(const unsigned*)&BsT[nl1][kb];
      b1[2 * v]     = us2bf((unsigned short)(p1 & 0xFFFFu));
      b1[2 * v + 1] = us2bf((unsigned short)(p1 >> 16));
    }
    acc0 = __builtin_amdgcn_wmma_f32_16x16x32_bf16(false, a, false, b0, (short)0, acc0, false, false);
    acc1 = __builtin_amdgcn_wmma_f32_16x16x32_bf16(false, a, false, b1, (short)0, acc1, false, false);
    __syncthreads();
  }

  // epilogue: bias + sigmoid, scatter to z[t][b][n] (t-major for the scan)
  const int mloc = tm * 16 + (lane >> 4) * 8;
#pragma unroll
  for (int r = 0; r < 8; ++r) {
    const int m = m0 + mloc + r;
    const int b = m >> 10;       // x rows are b*T + t, T=1024
    const int t = m & 1023;
    {
      const int n = n0 + tn0 * 16 + (lane & 15);
      const float vv = acc0[r] + b_w[n];
      z[((size_t)t * 64 + b) * 512 + n] = 1.0f / (1.0f + __expf(-vv));
    }
    {
      const int n = n0 + (tn0 + 1) * 16 + (lane & 15);
      const float vv = acc1[r] + b_w[n];
      z[((size_t)t * 64 + b) * 512 + n] = 1.0f / (1.0f + __expf(-vv));
    }
  }
}

// ---------------------------------------------------------------------------
// Kernel 3: the scan. One workgroup (32 waves) on one WGP. State in LDS:
//   cS : f32  [64][512] = 128 KB (accumulated in f32)
//   yB : bf16 [64][512] =  64 KB (row-major; WMMA operand precision)
// Per step: r = tanh(y@R + b_r) via bf16 WMMA (R pre-packed in B-fragment
// order -> pure global_load_b128), then c += i*z_t; y = o*tanh(c).
// A fragment = two ds_load_b128 per lane, zero conversions in the GEMM loop.
// Wave w owns columns [16w,16w+16) of both gate halves -> no cross-wave
// dependencies inside a step; two barriers/step order y reads vs y writes.
// ---------------------------------------------------------------------------
__global__ __launch_bounds__(1024) void lstm_scan_kernel(
    const float* __restrict__ z, const unsigned short* __restrict__ Rpk,
    const float* __restrict__ b_r, const float* __restrict__ y0,
    const float* __restrict__ c0, float* __restrict__ out) {
  extern __shared__ float smem[];
  float* cS = smem;                                  // [64][512] f32
  unsigned short* yB = (unsigned short*)(smem + 64 * 512);  // [64][512] bf16
  const int tid  = threadIdx.x;
  const int lane = tid & 31;
  const int wv   = tid >> 5;      // 0..31 == column tile (i-half)

  for (int idx = tid; idx < 64 * 512; idx += 1024) {
    cS[idx] = c0[idx];
    yB[idx] = f2bf_rne(y0[idx]);
  }

  const int   nloc = lane & 15;
  const int   n_i  = wv * 16 + nloc;          // 0..511
  const int   kh   = (lane >> 4) * 8;
  const float bri  = b_r[n_i];
  const float bro  = b_r[512 + n_i];
  v8f zero = {};

  for (int t = 0; t < 1024; ++t) {
    __syncthreads();              // prev-step y writes visible

    // prefetch next step's z tile for this wave (2 rows per thread)
    const int tnx = (t < 1023) ? t + 1 : t;
    __builtin_prefetch(&z[((size_t)tnx * 64 + lane) * 512 + wv * 16], 0, 0);
    __builtin_prefetch(&z[((size_t)tnx * 64 + 32 + lane) * 512 + wv * 16], 0, 0);

    v8f acc[4][2];
#pragma unroll
    for (int p = 0; p < 4; ++p) { acc[p][0] = zero; acc[p][1] = zero; }

    for (int kt = 0; kt < 16; ++kt) {
      const v16bf bi = *(const v16bf*)(Rpk + ((size_t)((kt * 64 + wv) * 32 + lane) * 16));
      const v16bf bo = *(const v16bf*)(Rpk + ((size_t)((kt * 64 + wv + 32) * 32 + lane) * 16));
#pragma unroll
      for (int p = 0; p < 4; ++p) {
        const int mrow = p * 16 + (lane & 15);
        const int base = mrow * 512 + kt * 32 + kh;   // ushort index
        v8bf lo = *(const v8bf*)(yB + base);          // k = k0+kh   .. +7
        v8bf hi = *(const v8bf*)(yB + base + 16);     // k = k0+16+kh.. +7
        v16bf a = __builtin_shufflevector(lo, hi, 0, 1, 2, 3, 4, 5, 6, 7,
                                          8, 9, 10, 11, 12, 13, 14, 15);
        acc[p][0] = __builtin_amdgcn_wmma_f32_16x16x32_bf16(false, a, false, bi, (short)0, acc[p][0], false, false);
        acc[p][1] = __builtin_amdgcn_wmma_f32_16x16x32_bf16(false, a, false, bo, (short)0, acc[p][1], false, false);
      }
    }
    __syncthreads();              // all y reads done before y writes

#pragma unroll
    for (int p = 0; p < 4; ++p) {
#pragma unroll
      for (int r = 0; r < 8; ++r) {
        const int    m  = p * 16 + r + (lane >> 4) * 8;
        const float  iv = tanhf(acc[p][0][r] + bri);
        const float  ov = tanhf(acc[p][1][r] + bro);
        const size_t gi = ((size_t)t * 64 + m) * 512 + n_i;
        const int    li = m * 512 + n_i;
        const float  cv = cS[li] + iv * z[gi];
        cS[li] = cv;
        const float  yv = ov * tanhf(cv);
        yB[li] = f2bf_rne(yv);    // state for next step (WMMA precision)
        out[gi] = yv;             // outputs[t][b][h] (f32, exact)
        if (t == 1023) out[(size_t)1024 * 64 * 512 + li] = yv;  // y_f
      }
    }
  }

  __syncthreads();
  const size_t TBH = (size_t)1024 * 64 * 512;
  for (int idx = tid; idx < 64 * 512; idx += 1024) {
    out[TBH + 32768 + idx] = cS[idx];   // c_f
  }
}

// ---------------------------------------------------------------------------
extern "C" void kernel_launch(void* const* d_in, const int* in_sizes, int n_in,
                              void* d_out, int out_size, void* d_ws, size_t ws_size,
                              hipStream_t stream) {
  const float* x   = (const float*)d_in[0];
  const float* W   = (const float*)d_in[1];
  const float* R   = (const float*)d_in[2];
  const float* b_r = (const float*)d_in[3];
  const float* b_w = (const float*)d_in[4];
  const float* y0  = (const float*)d_in[5];
  const float* c0  = (const float*)d_in[6];
  float* out = (float*)d_out;

  unsigned short* Rpk = (unsigned short*)d_ws;              // 1 MiB, fragment order
  float* z = (float*)((char*)d_ws + (1u << 20));            // 128 MiB, [T][B][H] f32

  pack_R_kernel<<<128, 256, 0, stream>>>(R, Rpk);

  dim3 grid(8, 1024);                                       // N tiles x M tiles (64x64)
  z_gemm_kernel<<<grid, 256, 0, stream>>>(x, W, b_w, z);

  const size_t shmem = 64 * 512 * sizeof(float)             // cS  (128 KiB)
                     + 64 * 512 * sizeof(unsigned short);   // yB  ( 64 KiB)
  hipFuncSetAttribute(reinterpret_cast<const void*>(lstm_scan_kernel),
                      hipFuncAttributeMaxDynamicSharedMemorySize, (int)shmem);
  lstm_scan_kernel<<<1, 1024, shmem, stream>>>(z, Rpk, b_r, y0, c0, out);
}
